// Painting_57243324121402
// MI455X (gfx1250) — compile-verified, matrix operand order checked by reading, not run on hardware
//
#include <hip/hip_runtime.h>
#include <hip/hip_bf16.h>
#include <math.h>

typedef __attribute__((ext_vector_type(16))) _Float16 v16h;
typedef __attribute__((ext_vector_type(8)))  float    v8f;

#define NS       512
#define HH       256
#define WW       256
#define NPIX     (HH*WW)
#define EPSV     1e-6f
#define LOG2E    1.4426950408889634f
#define LN2      0.6931471805599453f
#define PSTRIDE  16              // floats per stroke in the packed param table
#define WS_PARAM_OFF 64          // float offset (256 B) of param table in d_ws
#define A_OFF_BYTES  (256 + NS*PSTRIDE*4)   // byte offset of packed-A table
#define A_ELEMS      (16*32*16)  // chunks x lanes x elems = 8192 f16 (16 KB)
// ws[0] = l2 sum, ws[1] = sharpness sum, ws[2] = |alpha| sum

// Raw v_exp_f32 (2^x), no libm range-guard code. Arguments here are always
// <= 0 (qd >= 0), so the guarded path is unnecessary anyway.
__device__ __forceinline__ float fast_exp2(float x) {
#if __has_builtin(__builtin_amdgcn_exp2f)
    return __builtin_amdgcn_exp2f(x);
#else
    return __expf(x * LN2);
#endif
}

// ---------------------------------------------------------------------------
// Kernel 1: per-stroke max_pdf via analytic per-row minimization of the
// convex quadratic  quad = A*dx^2 - 2*B*dx*dy + C*dy^2:
// row minimizer x* = ox + (B/A)*dy; the discrete row min is at floor/ceil of
// the corresponding grid column (quad is unimodal along the row). One thread
// per row -> 2 evals instead of 256. Packs per-stroke params with log2(e)
// folded into the sigma terms so the paint kernel uses raw v_exp_f32.
// ---------------------------------------------------------------------------
__global__ void prep_kernel(const float* __restrict__ offset,
                            const float* __restrict__ sigma,
                            const float* __restrict__ theta,
                            const float* __restrict__ color,
                            const float* __restrict__ alpha,
                            float* __restrict__ ws) {
    __shared__ float smin[256];
    const int s   = blockIdx.x;
    const int tid = threadIdx.x;          // row index (HH == 256 == blockDim.x)

    const float ox = offset[s*2 + 0];
    const float oy = offset[s*2 + 1];
    const float sx = sigma[s*2 + 0];
    const float sy = sigma[s*2 + 1];
    const float th = theta[s];
    const float ct = __cosf(th);
    const float st = __sinf(th);

    const float ratio = (float)WW / (float)HH;      // 1.0 here, kept general
    const float sxe   = sx / ratio;
    const float i2x   = 1.0f / (2.0f * sxe * sxe);
    const float i2y   = 1.0f / (2.0f * sy  * sy );

    // quad(x,y) = Aq*dx^2 - 2*Bq*dx*dy + Cq*dy^2  (Aq > 0)
    const float Aq = i2x*ct*ct + i2y*st*st;
    const float Bq = ct*st*(i2x + i2y);

    const float y  = (float)tid * (1.0f/(float)HH);
    const float dy = y - oy;
    const float xstar = ox + (Bq / Aq) * dy;        // continuous row minimizer
    const float j0 = floorf(xstar * (float)WW);

    float mn = 3.4e38f;
#pragma unroll
    for (int t = 0; t < 2; ++t) {                   // floor & ceil candidates
        const float jc = fminf(fmaxf(j0 + (float)t, 0.0f), (float)(WW-1));
        const float x  = jc * (1.0f/(float)WW);
        const float dx = x - ox;
        const float cx = ct*dx - st*dy;             // rot = [[c,-s],[s,-c]]
        const float cy = st*dx - ct*dy;
        mn = fminf(mn, cx*cx*i2x + cy*cy*i2y);
    }
    smin[tid] = mn;
    __syncthreads();
    for (int off = 128; off > 0; off >>= 1) {
        if (tid < off) smin[tid] = fminf(smin[tid], smin[tid + off]);
        __syncthreads();
    }
    if (tid == 0) {
        const float maxpdf = fast_exp2(-smin[0] * LOG2E);
        const float invn   = 1.0f / (maxpdf + EPSV);
        const float al     = alpha[s];
        float* pp = ws + WS_PARAM_OFF + s * PSTRIDE;
        pp[0] = ox;            pp[1] = oy;
        pp[2] = ct;            pp[3] = st;
        pp[4] = i2x * LOG2E;   pp[5] = i2y * LOG2E;   // exp2-ready
        pp[6] = invn;          pp[7] = 0.0f;
        pp[8]  = al * color[s*3 + 0];
        pp[9]  = al * color[s*3 + 1];
        pp[10] = al * color[s*3 + 2];
        pp[11] = 0.0f; pp[12] = 0.0f; pp[13] = 0.0f; pp[14] = 0.0f; pp[15] = 0.0f;
        atomicAdd(ws + 1, fabsf(1.0f - sx / sy));   // sharpness term sum
        atomicAdd(ws + 2, fabsf(al));               // transparency term sum
    }
}

// ---------------------------------------------------------------------------
// Kernel 1b: pre-pack the WMMA A operand (alpha*color, rows 0-2; rows 3-15 = 0)
// in the exact per-lane VGPR striping of V_WMMA_F32_16X16X32_F16:
//   lane -> M = lane&15, half = lane>>4;  K(e) = 8*half + e + (e>=8 ? 8 : 0)
// aTab[((chunk*32 + lane)*16 + e)]  — 16 f16 (32 B) per (chunk,lane).
// ---------------------------------------------------------------------------
__global__ void packA_kernel(const float* __restrict__ color,
                             const float* __restrict__ alpha,
                             _Float16* __restrict__ aTab) {
    const int idx  = blockIdx.x * blockDim.x + threadIdx.x;   // 0..8191
    const int e    = idx & 15;
    const int lane = (idx >> 4) & 31;
    const int c    = idx >> 9;
    const int m    = lane & 15;
    const int half = lane >> 4;
    const int k    = c * 32 + half * 8 + e + ((e >= 8) ? 8 : 0);
    const float v  = (m < 3) ? alpha[k] * color[k*3 + m] : 0.0f;
    aTab[idx] = (_Float16)v;
}

// ---------------------------------------------------------------------------
// Kernel 2: fused gaussian-eval + WMMA composite + sigmoid + L2 partial.
// Per wave: 16 pixels (N), K = 512 strokes as 16 chunks of 32.
// A: pre-packed table, 2x ds_load_b128, branch-free. B: generated in regs
// (raw v_exp_f32 + v_cvt_pk_f16_f32). EXEC stays all-ones.
// ---------------------------------------------------------------------------
__global__ void paint_kernel(const float* __restrict__ canvas,
                             const float* __restrict__ target,
                             float* __restrict__ out,
                             float* __restrict__ ws) {
    __shared__ float sparams[NS * PSTRIDE];                 // 32 KB stroke table
    __shared__ __align__(32) _Float16 sA[A_ELEMS];          // 16 KB packed A
    __shared__ float red[256];

    const int tid = threadIdx.x;

    // cooperative stage of the stroke table + packed-A table (b128 copies)
    {
        const float4* src = (const float4*)(ws + WS_PARAM_OFF);
        float4*       dst = (float4*)sparams;
        for (int i = tid; i < NS * PSTRIDE / 4; i += blockDim.x) dst[i] = src[i];
        const float4* asrc = (const float4*)((const char*)ws + A_OFF_BYTES);
        float4*       adst = (float4*)sA;
        for (int i = tid; i < A_ELEMS / 8; i += blockDim.x) adst[i] = asrc[i];
    }
    __syncthreads();

    const int wave = tid >> 5;
    const int lane = tid & 31;
    const int half = lane >> 4;
    const int m    = lane & 15;                              // N col for B/D
    const int p    = blockIdx.x * 128 + wave * 16 + m;       // this lane's pixel
    const float x  = (float)(p & (WW-1)) * (1.0f/(float)WW);
    const float y  = (float)(p >> 8)     * (1.0f/(float)HH);

    const float4* pv = (const float4*)sparams;

    v8f acc = {};
#pragma unroll 2
    for (int c = 0; c < 16; ++c) {                           // 16 chunks of K=32
        const int kbase = c * 32 + half * 8;
        const v16h A = *(const v16h*)(&sA[(c * 32 + lane) * 16]);
        v16h B;
#pragma unroll
        for (int e = 0; e < 16; ++e) {
            const int k  = kbase + e + ((e >= 8) ? 8 : 0);
            const float4 q0 = pv[k*4 + 0];                   // ox, oy, cos, sin
            const float4 q1 = pv[k*4 + 1];                   // i2x*log2e, i2y*log2e, invnorm, -
            const float dx = x - q0.x, dy = y - q0.y;
            const float cx = q0.z*dx - q0.w*dy;
            const float cy = q0.w*dx - q0.z*dy;
            const float qd = cx*cx*q1.x + cy*cy*q1.y;        // in log2 units
            const float b  = fast_exp2(-qd) * q1.z;          // normalized pdf in [0,1]
            B[e] = (_Float16)b;
        }
        acc = __builtin_amdgcn_wmma_f32_16x16x32_f16(
                  /*neg_a=*/false, A, /*neg_b=*/false, B,
                  /*c_mod=*/(short)0, acc, /*reuse_a=*/false, /*reuse_b=*/false);
    }

    // D layout: lanes 0-15 hold rows M=0..7 in acc[0..7]; channels are rows 0-2.
    float l2 = 0.0f;
    if (half == 0) {
#pragma unroll
        for (int ch = 0; ch < 3; ++ch) {
            const float v = canvas[ch*NPIX + p] + acc[ch];
            const float o = 1.0f / (1.0f + fast_exp2(-v * LOG2E));   // sigmoid
            out[ch*NPIX + p] = o;
            const float d = o - target[ch*NPIX + p];
            l2 += d * d;
        }
    }
    red[tid] = l2;
    __syncthreads();
    for (int off = 128; off > 0; off >>= 1) {
        if (tid < off) red[tid] += red[tid + off];
        __syncthreads();
    }
    if (tid == 0) atomicAdd(ws, red[0]);
}

// ---------------------------------------------------------------------------
// Kernel 3: scalar loss
// ---------------------------------------------------------------------------
__global__ void finalize_kernel(const float* __restrict__ ws,
                                float* __restrict__ out) {
    if (threadIdx.x == 0 && blockIdx.x == 0) {
        const float l2    = ws[0] / (float)(3 * NPIX);
        const float sharp = (ws[1] / (float)NS) * 0.001f;
        const float trans = -(ws[2] / (float)NS) * 0.001f;
        const float mse   = l2;
        const float psnr  = 10.0f * log10f(1.0f / (mse + 1e-12f));
        out[3 * NPIX] = l2 + trans + sharp - psnr / 30.0f;
    }
}

// ---------------------------------------------------------------------------
extern "C" void kernel_launch(void* const* d_in, const int* in_sizes, int n_in,
                              void* d_out, int out_size, void* d_ws, size_t ws_size,
                              hipStream_t stream) {
    const float* canvas = (const float*)d_in[0];
    const float* target = (const float*)d_in[1];
    const float* offset = (const float*)d_in[2];
    const float* sigma  = (const float*)d_in[3];
    const float* theta  = (const float*)d_in[4];
    const float* color  = (const float*)d_in[5];
    const float* alpha  = (const float*)d_in[6];
    float* out = (float*)d_out;
    float* ws  = (float*)d_ws;
    _Float16* aTab = (_Float16*)((char*)d_ws + A_OFF_BYTES);

    // zero loss accumulators (graph-capturable memset node)
    hipMemsetAsync(d_ws, 0, 256, stream);

    prep_kernel<<<NS, 256, 0, stream>>>(offset, sigma, theta, color, alpha, ws);
    packA_kernel<<<A_ELEMS / 256, 256, 0, stream>>>(color, alpha, aTab);
    paint_kernel<<<NPIX / 128, 256, 0, stream>>>(canvas, target, out, ws);
    finalize_kernel<<<1, 64, 0, stream>>>(ws, out);
}
